// NeuralAdditiveModel_75127567941662
// MI455X (gfx1250) — matrix operand order repaired and checked by reading, not compiled, not run on hardware
//
#include <hip/hip_runtime.h>

// NAM forward for MI455X (gfx1250, wave32, WMMA).
// B=4096, F=512, S=32, H1=64, H2=32.
#define B_TOT 4096
#define F_TOT 512

typedef _Float16 f16x16 __attribute__((ext_vector_type(16)));
typedef _Float16 f16x8  __attribute__((ext_vector_type(8)));
typedef float    f32x8  __attribute__((ext_vector_type(8)));

// One block = one feature f (blockIdx.x) x 256 batch rows (blockIdx.y).
// 256 threads = 8 waves; wave w owns rows {tile + 16w + 128*rt : rt in 0,1}.
__global__ __launch_bounds__(256) void nam_main(
    const float* __restrict__ xg,   // (B,F)
    const float* __restrict__ W0g,  // (F,32,32)
    const float* __restrict__ b0g,  // (F,32)
    const float* __restrict__ W1g,  // (F,32,64)
    const float* __restrict__ b1g,  // (F,32)
    const float* __restrict__ W2g,  // (F,64,32)
    const float* __restrict__ b2g,  // (F,64)
    const float* __restrict__ Wlg,  // (F,32)
    float* __restrict__ fout)       // (B,F)
{
    __shared__ float sc0[32], sd0[32], se1[64], se2[32], sWl[32];
    __shared__ _Float16 sW1[64 * 32];      // transposed: [h][s], s contiguous
    __shared__ _Float16 sW2[32 * 64];      // transposed: [o][h], h contiguous
    __shared__ _Float16 sh1[8][16 * 72];   // per-wave 16x64 h1 tile, padded rows

    const int f   = blockIdx.x;
    const int tid = threadIdx.x;

    // ---- per-feature precompute (rank-1 folding of ExU layer + bias@W) ----
    if (tid < 32) {
        float c = 0.f, d = 0.f;
        const float* w0 = W0g + f * 1024 + tid;   // stride 32 over s
        const float* b0 = b0g + f * 32;
        for (int s = 0; s < 32; ++s) { float w = __expf(w0[s * 32]); c += w; d += b0[s] * w; }
        sc0[tid] = c; sd0[tid] = d;
    } else if (tid < 96) {
        int h = tid - 32;
        float e = 0.f;
        const float* w1 = W1g + f * 2048 + h;     // stride 64 over s
        const float* b1 = b1g + f * 32;
        for (int s = 0; s < 32; ++s) e += b1[s] * w1[s * 64];
        se1[h] = e;
    } else if (tid < 128) {
        int o = tid - 96;
        float e = 0.f;
        const float* w2 = W2g + f * 2048 + o;     // stride 32 over h
        const float* b2 = b2g + f * 64;
        for (int h = 0; h < 64; ++h) e += b2[h] * w2[h * 32];
        se2[o] = e;
    } else if (tid < 160) {
        sWl[tid - 128] = Wlg[f * 32 + (tid - 128)];
    }
    // ---- weights -> f16 LDS, K-contiguous so a B fragment is one v16 load ----
    #pragma unroll
    for (int i = 0; i < 8; ++i) {
        int idx = tid + i * 256;                          // 0..2047
        int h = idx >> 5, s = idx & 31;
        sW1[idx] = (_Float16)W1g[f * 2048 + s * 64 + h];  // sW1[h*32+s]
        int o = idx >> 6, hh = idx & 63;
        sW2[idx] = (_Float16)W2g[f * 2048 + hh * 32 + o]; // sW2[o*64+hh]
    }
    __syncthreads();

    const int lane = tid & 31;
    const int wave = tid >> 5;
    const int half = lane >> 4;     // A/B layout half (K-group)
    const int lrow = lane & 15;     // A row / B-C-D column within tile
    _Float16* myh1 = &sh1[wave][0];

    // ---- hoist all B fragments into registers (reused across row tiles) ----
    f16x16 bW1[4];
    #pragma unroll
    for (int nt = 0; nt < 4; ++nt)
        bW1[nt] = *(const f16x16*)&sW1[(nt * 16 + lrow) * 32 + half * 16];
    f16x16 bW2[2][2];
    #pragma unroll
    for (int nt = 0; nt < 2; ++nt)
        #pragma unroll
        for (int kk = 0; kk < 2; ++kk)
            bW2[kk][nt] = *(const f16x16*)&sW2[(nt * 16 + lrow) * 64 + kk * 32 + half * 16];

    #pragma unroll
    for (int rt = 0; rt < 2; ++rt) {
        const int rowBase = blockIdx.y * 256 + rt * 128 + wave * 16;
        const float xv = xg[(rowBase + lrow) * F_TOT + f];

        // ---- stage 0 (rank-1 ExU) directly in A-fragment layout ----
        f16x16 a0;
        #pragma unroll
        for (int e = 0; e < 16; ++e) {
            int k = half * 16 + e;
            float v = xv * sc0[k] - sd0[k];
            v = fminf(fmaxf(v, 0.f), 1.f);            // -> v_max ... clamp
            a0[e] = (_Float16)v;
        }

        // ---- stage 1: (16x32) @ (32x64), bias folded into C init ----
        #pragma unroll
        for (int nt = 0; nt < 4; ++nt) {
            int col = nt * 16 + lrow;
            f32x8 c;
            float ci = -se1[col];
            #pragma unroll
            for (int r = 0; r < 8; ++r) c[r] = ci;
            c = __builtin_amdgcn_wmma_f32_16x16x32_f16(false, a0, false, bW1[nt],
                                                       (short)0, c, false, false);
            #pragma unroll
            for (int r = 0; r < 8; ++r) {
                float v = c[r] > 0.f ? c[r] : 0.f;               // relu (select)
                myh1[(r + 8 * half) * 72 + col] = (_Float16)v;   // C-layout -> LDS
            }
        }
        __syncthreads();   // order per-wave LDS writes before A-layout reload

        // ---- reload h1 as A fragments (two K-splits of 32) ----
        f16x16 a2[2];
        #pragma unroll
        for (int kk = 0; kk < 2; ++kk) {
            const _Float16* base = myh1 + lrow * 72 + kk * 32 + half * 16;
            f16x8 lo = *(const f16x8*)base;          // elements 0..7  (16B aligned)
            f16x8 hi = *(const f16x8*)(base + 8);    // elements 8..15
            a2[kk] = __builtin_shufflevector(lo, hi, 0, 1, 2, 3, 4, 5, 6, 7,
                                             8, 9, 10, 11, 12, 13, 14, 15);
        }
        __syncthreads();   // sh1 reads done before next iteration overwrites

        // ---- stage 2: (16x64) @ (64x32) + relu + dot with Wl ----
        float p[8];
        #pragma unroll
        for (int r = 0; r < 8; ++r) p[r] = 0.f;

        #pragma unroll
        for (int nt = 0; nt < 2; ++nt) {
            int col = nt * 16 + lrow;
            f32x8 c;
            float ci = -se2[col];
            #pragma unroll
            for (int r = 0; r < 8; ++r) c[r] = ci;
            #pragma unroll
            for (int kk = 0; kk < 2; ++kk)
                c = __builtin_amdgcn_wmma_f32_16x16x32_f16(false, a2[kk], false, bW2[kk][nt],
                                                           (short)0, c, false, false);
            float wl = sWl[col];
            #pragma unroll
            for (int r = 0; r < 8; ++r) {
                float v = c[r] > 0.f ? c[r] : 0.f;               // relu (select)
                p[r] += v * wl;
            }
        }

        // ---- reduce each row across the 16 lanes of its half (wave32) ----
        #pragma unroll
        for (int r = 0; r < 8; ++r) {
            float v = p[r];
            v += __shfl_xor(v, 1, 32);
            v += __shfl_xor(v, 2, 32);
            v += __shfl_xor(v, 4, 32);
            v += __shfl_xor(v, 8, 32);
            p[r] = v;
        }
        if (lrow == 0) {
            #pragma unroll
            for (int r = 0; r < 8; ++r) {
                int b = rowBase + 8 * half + r;
                fout[b * F_TOT + f] = p[r];
            }
        }
    }
}

// out[b] = sum_f fout[b,f] + out_bias. One wave per row, coalesced reads.
__global__ __launch_bounds__(256) void nam_reduce(const float* __restrict__ fout,
                                                  const float* __restrict__ out_bias,
                                                  float* __restrict__ out)
{
    int row  = blockIdx.x * 8 + (threadIdx.x >> 5);
    int lane = threadIdx.x & 31;
    const float* rp = fout + row * F_TOT;
    float s = 0.f;
    #pragma unroll
    for (int i = 0; i < 16; ++i) s += rp[lane + i * 32];
    s += __shfl_xor(s, 16, 32);
    s += __shfl_xor(s, 8, 32);
    s += __shfl_xor(s, 4, 32);
    s += __shfl_xor(s, 2, 32);
    s += __shfl_xor(s, 1, 32);
    if (lane == 0) out[row] = s + out_bias[0];
}

extern "C" void kernel_launch(void* const* d_in, const int* in_sizes, int n_in,
                              void* d_out, int out_size, void* d_ws, size_t ws_size,
                              hipStream_t stream) {
    const float* x  = (const float*)d_in[0];
    const float* W0 = (const float*)d_in[1];
    const float* b0 = (const float*)d_in[2];
    const float* W1 = (const float*)d_in[3];
    const float* b1 = (const float*)d_in[4];
    const float* W2 = (const float*)d_in[5];
    const float* b2 = (const float*)d_in[6];
    const float* Wl = (const float*)d_in[7];
    const float* ob = (const float*)d_in[8];

    float* out  = (float*)d_out;     // first B_TOT elements: per-sample sums
    float* fout = out + B_TOT;       // then (B,F) per-feature outputs

    dim3 grid(F_TOT, B_TOT / 256);   // 512 features x 16 batch tiles
    nam_main<<<grid, 256, 0, stream>>>(x, W0, b0, W1, b1, W2, b2, Wl, fout);
    nam_reduce<<<B_TOT / 8, 256, 0, stream>>>(fout, ob, out);
}
